// Model_v6_2954937499688
// MI455X (gfx1250) — matrix operand order repaired and checked by reading, not compiled, not run on hardware
//
#include <hip/hip_runtime.h>
#include <hip/hip_bf16.h>

#define LSEQ   384
#define BATCH  8
#define HIDDEN 512
#define NHEADS 8
#define DKH    64
#define NL     4
#define MROWS  (BATCH*LSEQ)       // 3072
#define BH     (BATCH*NHEADS)     // 64
#define INF    __builtin_inff()

typedef __attribute__((ext_vector_type(16))) __bf16 v16bf;
typedef __attribute__((ext_vector_type(8)))  float  v8f;

#define DEV __device__ __forceinline__

DEV v8f wmma_bf16(v16bf a, v16bf b, v8f c) {
  return __builtin_amdgcn_wmma_f32_16x16x32_bf16(false, a, false, b, (short)0, c, false, false);
}

// A fragment (16x32 bf16, ISA 7.12.2): lane's 16 elements are two contiguous
// 8-element runs: k = 8*half + [0,8) and k = 16 + 8*half + [0,8)
DEV void load_a_frag(const __bf16* p, int half, v16bf& a) {
  const __bf16* p0 = p + 8 * half;
  const __bf16* p1 = p + 16 + 8 * half;
#pragma unroll
  for (int e = 0; e < 8; ++e) { a[e] = p0[e]; a[e + 8] = p1[e]; }
}
// B fragment (32x16 bf16): lane n's 16 elements are two 8-element runs:
// k = 8*half + [0,8) and k = 16 + 8*half + [0,8) (operand stored n-major / k-contiguous).
DEV void load_b_frag(const __bf16* p, int half, v16bf& b) {
  const __bf16* q0 = p + 8 * half;
  const __bf16* q1 = p + 16 + 8 * half;
#pragma unroll
  for (int e = 0; e < 8; ++e) { b[e] = q0[e]; b[e + 8] = q1[e]; }
}

// ---------------------------------------------------------------- prep
__global__ void k_prep(const float* __restrict__ x, const int* __restrict__ mask,
                       float* __restrict__ h, __bf16* __restrict__ hbf) {
  int idx = blockIdx.x * 256 + threadIdx.x;           // over MROWS*HIDDEN
  int m = idx >> 9;
  float v = mask[m] ? x[idx] : 0.0f;
  h[idx] = v;
  hbf[idx] = (__bf16)v;
}

// ---------------------------------------------------------------- GEMM [3072x512] x [512x512]
// QKV mode: outf==nullptr, blockIdx.z in {0,1,2}; z==2 stores V transposed [B,HIDDEN,LSEQ]
// O mode:   outf!=nullptr, epilogue outf = resid + acc + bias (f32)
__global__ void k_gemm512(const __bf16* __restrict__ Abf,
                          const float* __restrict__ W0, const float* __restrict__ W1,
                          const float* __restrict__ W2,
                          const float* __restrict__ b0, const float* __restrict__ b1,
                          const float* __restrict__ b2,
                          __bf16* __restrict__ O0, __bf16* __restrict__ O1,
                          __bf16* __restrict__ Vt,
                          const float* __restrict__ resid, float* __restrict__ outf) {
  __shared__ __align__(16) __bf16 As[64 * 32];        // row-major (A is k-contiguous already)
  __shared__ __align__(16) __bf16 Bst[32 * 32];       // TRANSPOSED: [n][k], k-contiguous

  int z = blockIdx.z;
  const float* W    = (outf != nullptr) ? W0 : (z == 0 ? W0 : (z == 1 ? W1 : W2));
  const float* bias = (outf != nullptr) ? b0 : (z == 0 ? b0 : (z == 1 ? b1 : b2));

  int mBase = blockIdx.x * 64;
  int nBase = blockIdx.y * 32;
  int tid = threadIdx.x, w = tid >> 5, lane = tid & 31;
  int half = lane >> 4, lr = lane & 15;
  int wm = w & 3, wn = w >> 2;

  int ar = tid >> 2, ac = (tid & 3) * 8;              // A stage: 64 x 32, 16B/thread
  int br = tid >> 3, bc = (tid & 7) * 4;              // B stage: 32 x 32, 4 f32/thread

  unsigned ldsA = (unsigned)(size_t)(As + ar * 32 + ac);   // wave-relative LDS byte addr

  v8f acc = {};
  for (int kc = 0; kc < 16; ++kc) {
    int k0 = kc * 32;
    // ---- stage A: async global->LDS copy (ASYNCcnt-tracked, no VGPR round trip)
    const __bf16* gptr = Abf + (size_t)(mBase + ar) * HIDDEN + k0 + ac;
    asm volatile("global_load_async_to_lds_b128 %0, %1, off"
                 :: "v"(ldsA), "v"((unsigned long long)(size_t)gptr) : "memory");
    // ---- stage B: f32 -> bf16, transposed to [n][k]
    const float4 wv = *(const float4*)(W + (size_t)(k0 + br) * HIDDEN + nBase + bc);
    Bst[(bc + 0) * 32 + br] = (__bf16)wv.x;
    Bst[(bc + 1) * 32 + br] = (__bf16)wv.y;
    Bst[(bc + 2) * 32 + br] = (__bf16)wv.z;
    Bst[(bc + 3) * 32 + br] = (__bf16)wv.w;
    asm volatile("s_wait_asynccnt 0" ::: "memory");
    __syncthreads();

    v16bf a, b;
    load_a_frag(As + (wm * 16 + lr) * 32, half, a);
    load_b_frag(Bst + (wn * 16 + lr) * 32, half, b);
    acc = wmma_bf16(a, b, acc);
    __syncthreads();
  }

  if (outf) {                                         // O projection + residual (f32)
#pragma unroll
    for (int r = 0; r < 8; ++r) {
      int m = mBase + wm * 16 + half * 8 + r;
      int n = nBase + wn * 16 + lr;
      outf[(size_t)m * HIDDEN + n] = resid[(size_t)m * HIDDEN + n] + acc[r] + bias[n];
    }
  } else if (z == 2) {                                // V: store transposed [B, HIDDEN, LSEQ]
    int bb = mBase / LSEQ;                            // 64 | 384 -> tile within one batch
    int j0 = (mBase - bb * LSEQ) + wm * 16 + half * 8;
    int n = nBase + wn * 16 + lr;
    union { __bf16 h[8]; uint4 q; } tmp;
#pragma unroll
    for (int r = 0; r < 8; ++r) tmp.h[r] = (__bf16)(acc[r] + bias[n]);
    *(uint4*)(Vt + ((size_t)(bb * HIDDEN + n)) * LSEQ + j0) = tmp.q;
  } else {                                            // Q / K row-major bf16
    __bf16* Obf = (z == 0) ? O0 : O1;
#pragma unroll
    for (int r = 0; r < 8; ++r) {
      int m = mBase + wm * 16 + half * 8 + r;
      int n = nBase + wn * 16 + lr;
      Obf[(size_t)m * HIDDEN + n] = (__bf16)(acc[r] + bias[n]);
    }
  }
}

// ---------------------------------------------------------------- pair-bias MLP 53->32->8
__global__ void k_pairbias(const float* __restrict__ refCov,
                           const float* __restrict__ Wr1, const float* __restrict__ br1,
                           const float* __restrict__ Wr2, const float* __restrict__ br2,
                           const int* __restrict__ mask, float* __restrict__ refb) {
  __shared__ __align__(16) __bf16 W1t[32 * 64];       // TRANSPOSED [n=32][k=64] (K pad 53->64)
  __shared__ __align__(16) __bf16 W2t[16 * 32];       // TRANSPOSED [n=16][k=32] (N pad 8->16)
  __shared__ __align__(16) __bf16 hid[8][16 * 32];    // per-wave transpose staging

  int tid = threadIdx.x;
  for (int idx = tid; idx < 32 * 64; idx += 256) {
    int n = idx >> 6, k = idx & 63;
    W1t[idx] = (__bf16)(k < 53 ? Wr1[k * 32 + n] : 0.0f);
  }
  for (int idx = tid; idx < 16 * 32; idx += 256) {
    int n = idx >> 5, k = idx & 31;
    W2t[idx] = (__bf16)(n < 8 ? Wr2[k * 8 + n] : 0.0f);
  }
  __syncthreads();

  int w = tid >> 5, lane = tid & 31, half = lane >> 4, lr = lane & 15;
  long m0 = (long)blockIdx.x * 128 + w * 16;          // flattened (b,i,j) row
  const float* arow = refCov + (size_t)(m0 + lr) * 53;

  v8f acc[2] = {{}, {}};
#pragma unroll
  for (int kc = 0; kc < 2; ++kc) {
    v16bf a;
#pragma unroll
    for (int e = 0; e < 8; ++e) {
      int k0 = kc * 32 + 8 * half + e;
      int k1 = kc * 32 + 16 + 8 * half + e;
      a[e]     = (__bf16)(k0 < 53 ? arow[k0] : 0.0f);
      a[e + 8] = (__bf16)(k1 < 53 ? arow[k1] : 0.0f);
    }
#pragma unroll
    for (int t = 0; t < 2; ++t) {
      v16bf b;
      load_b_frag(W1t + (t * 16 + lr) * 64 + kc * 32, half, b);
      acc[t] = wmma_bf16(a, b, acc[t]);
    }
  }

  // bias + relu, stage through LDS to re-layout C(16x32) -> A fragment
#pragma unroll
  for (int t = 0; t < 2; ++t)
#pragma unroll
    for (int r = 0; r < 8; ++r) {
      int n = t * 16 + lr;
      float hv = fmaxf(acc[t][r] + br1[n], 0.0f);
      hid[w][(half * 8 + r) * 32 + n] = (__bf16)hv;
    }
  asm volatile("s_wait_dscnt 0" ::: "memory");        // wave-local LDS RAW fence

  v16bf a2, b2;
  load_a_frag(hid[w] + lr * 32, half, a2);
  load_b_frag(W2t + lr * 32, half, b2);
  v8f zc = {};
  v8f out = wmma_bf16(a2, b2, zc);

  int hd = lr;                                        // head (valid < 8)
  if (hd < 8) {
#pragma unroll
    for (int r = 0; r < 8; ++r) {
      long m = m0 + half * 8 + r;
      int j = (int)(m % LSEQ);
      long t2 = m / LSEQ;
      int i = (int)(t2 % LSEQ);
      int b = (int)(t2 / LSEQ);
      bool mv = mask[b * LSEQ + i] && mask[b * LSEQ + j];
      refb[((size_t)(b * NHEADS + hd) * LSEQ + i) * LSEQ + j] = mv ? (out[r] + br2[hd]) : -INF;
    }
  }
}

// ---------------------------------------------------------------- scores = Q K^T / 8 (masked)
__global__ void k_scores(const __bf16* __restrict__ Qbf, const __bf16* __restrict__ Kbf,
                         const int* __restrict__ mask, float* __restrict__ scores) {
  int z = blockIdx.z, b = z >> 3, hd = z & 7, h0 = hd * DKH;
  int iBase = blockIdx.x * 64, jBase = blockIdx.y * 64;
  int tid = threadIdx.x, w = tid >> 5, lane = tid & 31;
  int half = lane >> 4, lr = lane & 15;
  int wm = w & 3, wn = w >> 2;

  int i = iBase + wm * 16 + lr;
  const __bf16* qrow = Qbf + ((size_t)(b * LSEQ + i)) * HIDDEN + h0;

  v8f acc[2] = {{}, {}};
#pragma unroll
  for (int kc = 0; kc < 2; ++kc) {
    v16bf a;
    load_a_frag(qrow + kc * 32, half, a);             // K^T: B's k = d, contiguous in K row
#pragma unroll
    for (int t = 0; t < 2; ++t) {
      int j = jBase + wn * 32 + t * 16 + lr;
      const __bf16* krow = Kbf + ((size_t)(b * LSEQ + j)) * HIDDEN + h0;
      v16bf bb;
      load_b_frag(krow + kc * 32, half, bb);
      acc[t] = wmma_bf16(a, bb, acc[t]);
    }
  }

#pragma unroll
  for (int t = 0; t < 2; ++t)
#pragma unroll
    for (int r = 0; r < 8; ++r) {
      int ii = iBase + wm * 16 + half * 8 + r;
      int jj = jBase + wn * 32 + t * 16 + lr;
      bool mv = mask[b * LSEQ + ii] && mask[b * LSEQ + jj];
      scores[((size_t)z * LSEQ + ii) * LSEQ + jj] = mv ? acc[t][r] * 0.125f : -INF;
    }
}

// ---------------------------------------------------------------- dual softmax + combine
DEV float redMax(float v, float* red, int tid) {
  red[tid] = v; __syncthreads();
  for (int s = 64; s > 0; s >>= 1) { if (tid < s) red[tid] = fmaxf(red[tid], red[tid + s]); __syncthreads(); }
  float r = red[0]; __syncthreads(); return r;
}
DEV float redSum(float v, float* red, int tid) {
  red[tid] = v; __syncthreads();
  for (int s = 64; s > 0; s >>= 1) { if (tid < s) red[tid] += red[tid + s]; __syncthreads(); }
  float r = red[0]; __syncthreads(); return r;
}

__global__ void k_softmax(const float* __restrict__ scores, const float* __restrict__ refb,
                          __bf16* __restrict__ combbf) {
  __shared__ float red[128];
  size_t row = blockIdx.x;
  const float* ps = scores + row * LSEQ;
  const float* pr = refb + row * LSEQ;
  __bf16* pc = combbf + row * LSEQ;
  int tid = threadIdx.x;

  float ms = -INF, mr = -INF;
  for (int j = tid; j < LSEQ; j += 128) { ms = fmaxf(ms, ps[j]); mr = fmaxf(mr, pr[j]); }
  ms = redMax(ms, red, tid);
  mr = redMax(mr, red, tid);
  bool allS = (ms == -INF), allR = (mr == -INF);      // fully masked row -> uniform

  float ss = 0.0f, sr = 0.0f;
  for (int j = tid; j < LSEQ; j += 128) {
    if (!allS) ss += __expf(ps[j] - ms);
    if (!allR) sr += __expf(pr[j] - mr);
  }
  ss = redSum(ss, red, tid);
  sr = redSum(sr, red, tid);
  float is = allS ? 0.0f : 1.0f / ss;
  float ir = allR ? 0.0f : 1.0f / sr;
  float u = 1.0f / (float)LSEQ;

  for (int j = tid; j < LSEQ; j += 128) {
    float aw = allS ? u : __expf(ps[j] - ms) * is;
    float rw = allR ? u : __expf(pr[j] - mr) * ir;
    pc[j] = (__bf16)(0.5f * (aw + rw));               // each dist sums to 1 -> /2
  }
}

// ---------------------------------------------------------------- out = comb @ V  (V transposed)
__global__ void k_attnout(const __bf16* __restrict__ combbf, const __bf16* __restrict__ Vt,
                          __bf16* __restrict__ attnbf) {
  int z = blockIdx.z, b = z >> 3, hd = z & 7, h0 = hd * DKH;
  int iBase = blockIdx.x * 32;
  int tid = threadIdx.x, w = tid >> 5, lane = tid & 31;
  int half = lane >> 4, lr = lane & 15;
  int wm = w & 1, wn = w >> 1;                        // 2 x 4 wave grid: 32 rows x 64 cols

  int i = iBase + wm * 16 + lr;
  const __bf16* arow = combbf + ((size_t)z * LSEQ + i) * LSEQ;
  int d = wn * 16 + lr;
  const __bf16* vrow = Vt + ((size_t)(b * HIDDEN + h0 + d)) * LSEQ;  // k = j contiguous

  v8f acc = {};
  for (int jc = 0; jc < 12; ++jc) {
    v16bf a, bb;
    load_a_frag(arow + jc * 32, half, a);
    load_b_frag(vrow + jc * 32, half, bb);
    acc = wmma_bf16(a, bb, acc);
  }
#pragma unroll
  for (int r = 0; r < 8; ++r) {
    int ii = iBase + wm * 16 + half * 8 + r;
    attnbf[((size_t)(b * LSEQ + ii)) * HIDDEN + h0 + wn * 16 + lr] = (__bf16)acc[r];
  }
}

// ---------------------------------------------------------------- layernorm (row = 512)
__global__ void k_layernorm(const float* __restrict__ in, const float* __restrict__ g,
                            const float* __restrict__ be, float* __restrict__ out,
                            __bf16* __restrict__ outbf) {
  __shared__ float red[256];
  size_t row = blockIdx.x;
  int tid = threadIdx.x;
  const float* p = in + row * HIDDEN;
  float v0 = p[tid], v1 = p[tid + 256];

  red[tid] = v0 + v1; __syncthreads();
  for (int s = 128; s > 0; s >>= 1) { if (tid < s) red[tid] += red[tid + s]; __syncthreads(); }
  float mu = red[0] * (1.0f / HIDDEN); __syncthreads();

  float d0 = v0 - mu, d1 = v1 - mu;
  red[tid] = d0 * d0 + d1 * d1; __syncthreads();
  for (int s = 128; s > 0; s >>= 1) { if (tid < s) red[tid] += red[tid + s]; __syncthreads(); }
  float rs = rsqrtf(red[0] * (1.0f / HIDDEN) + 1e-5f);

  float o0 = d0 * rs * g[tid] + be[tid];
  float o1 = d1 * rs * g[tid + 256] + be[tid + 256];
  out[row * HIDDEN + tid] = o0;
  out[row * HIDDEN + tid + 256] = o1;
  if (outbf) {
    outbf[row * HIDDEN + tid] = (__bf16)o0;
    outbf[row * HIDDEN + tid + 256] = (__bf16)o1;
  }
}

// ---------------------------------------------------------------- masked sum over L
__global__ void k_reduce(const float* __restrict__ h, const int* __restrict__ mask,
                         float* __restrict__ out) {
  int b = blockIdx.x, c = threadIdx.x;                // 8 blocks x 512 threads
  float acc = 0.0f;
  for (int l = 0; l < LSEQ; ++l)
    if (mask[b * LSEQ + l]) acc += h[((size_t)(b * LSEQ + l)) * HIDDEN + c];
  out[b * HIDDEN + c] = acc;
}

// ================================================================ launch
extern "C" void kernel_launch(void* const* d_in, const int* in_sizes, int n_in,
                              void* d_out, int out_size, void* d_ws, size_t ws_size,
                              hipStream_t stream) {
  const float* x      = (const float*)d_in[0];
  const int*   mask   = (const int*)  d_in[1];
  const float* refCov = (const float*)d_in[2];
  const float* Wq  = (const float*)d_in[3];  const float* bq  = (const float*)d_in[4];
  const float* Wk  = (const float*)d_in[5];  const float* bk  = (const float*)d_in[6];
  const float* Wv  = (const float*)d_in[7];  const float* bv  = (const float*)d_in[8];
  const float* Wo  = (const float*)d_in[9];  const float* bo  = (const float*)d_in[10];
  const float* Wr1 = (const float*)d_in[11]; const float* br1 = (const float*)d_in[12];
  const float* Wr2 = (const float*)d_in[13]; const float* br2 = (const float*)d_in[14];
  const float* ln_g = (const float*)d_in[15]; const float* ln_b = (const float*)d_in[16];
  const float* fn_g = (const float*)d_in[17]; const float* fn_b = (const float*)d_in[18];

  char* ws = (char*)d_ws;
  size_t off = 0;
  auto alloc = [&](size_t bytes) { void* p = ws + off; off += (bytes + 255) & ~(size_t)255; return p; };

  float*  h      = (float*) alloc((size_t)MROWS * HIDDEN * 4);
  float*  hnext  = (float*) alloc((size_t)MROWS * HIDDEN * 4);
  __bf16* hbf    = (__bf16*)alloc((size_t)MROWS * HIDDEN * 2);
  __bf16* Qbf    = (__bf16*)alloc((size_t)MROWS * HIDDEN * 2);
  __bf16* Kbf    = (__bf16*)alloc((size_t)MROWS * HIDDEN * 2);
  __bf16* Vt     = (__bf16*)alloc((size_t)MROWS * HIDDEN * 2);   // transposed V
  __bf16* attnbf = (__bf16*)alloc((size_t)MROWS * HIDDEN * 2);
  float*  scores = (float*) alloc((size_t)BH * LSEQ * LSEQ * 4);
  float*  refb   = (float*) alloc((size_t)BH * LSEQ * LSEQ * 4);
  __bf16* combbf = (__bf16*)alloc((size_t)BH * LSEQ * LSEQ * 2);

  k_prep<<<MROWS * HIDDEN / 256, 256, 0, stream>>>(x, mask, h, hbf);

  for (int i = 0; i < NL; ++i) {
    const size_t wOff = (size_t)i * HIDDEN * HIDDEN, bOff = (size_t)i * HIDDEN;
    k_gemm512<<<dim3(MROWS/64, HIDDEN/32, 3), 256, 0, stream>>>(
        hbf, Wq + wOff, Wk + wOff, Wv + wOff, bq + bOff, bk + bOff, bv + bOff,
        Qbf, Kbf, Vt, nullptr, nullptr);
    k_pairbias<<<(BATCH * LSEQ * LSEQ) / 128, 256, 0, stream>>>(
        refCov, Wr1 + (size_t)i * 53 * 32, br1 + (size_t)i * 32,
        Wr2 + (size_t)i * 32 * 8, br2 + (size_t)i * 8, mask, refb);
    k_scores<<<dim3(LSEQ/64, LSEQ/64, BH), 256, 0, stream>>>(Qbf, Kbf, mask, scores);
    k_softmax<<<BH * LSEQ, 128, 0, stream>>>(scores, refb, combbf);
    k_attnout<<<dim3(LSEQ/32, 1, BH), 256, 0, stream>>>(combbf, Vt, attnbf);
    k_gemm512<<<dim3(MROWS/64, HIDDEN/32, 1), 256, 0, stream>>>(
        attnbf, Wo + wOff, nullptr, nullptr, bo + bOff, nullptr, nullptr,
        nullptr, nullptr, nullptr, h, hnext);
    k_layernorm<<<MROWS, 256, 0, stream>>>(hnext, ln_g + bOff, ln_b + bOff, h, hbf);
  }

  k_layernorm<<<MROWS, 256, 0, stream>>>(h, fn_g, fn_b, h, nullptr);
  k_reduce<<<BATCH, HIDDEN, 0, stream>>>(h, mask, (float*)d_out);
}
